// FuncPitchEncoder_6176162972288
// MI455X (gfx1250) — matrix operand order repaired and checked by reading, not compiled, not run on hardware
//
#include <hip/hip_runtime.h>

// ---------------- constants (match reference) ----------------
#define LC      128      // pitch length
#define NCH     10       // conv channels
#define EMB     256
#define ZD      128
#define KCB     64       // codebook entries
#define FEAT    290      // NCH * 29 pooled features
#define KPAD    320      // FEAT padded to multiple of 32 for WMMA K
#define FSTRIDE 328      // LDS row stride (halves): 656B = 41*16B aligned, bank-conflict-free
#define ROWS    64       // batch rows per workgroup
#define MUSTR   132      // LDS stride for f32 mu

typedef __attribute__((ext_vector_type(16))) _Float16 v16h;
typedef __attribute__((ext_vector_type(8)))  _Float16 v8h;
typedef __attribute__((ext_vector_type(8)))  float    v8f;

// ---------------------------------------------------------------------------
// prep: Wc[n][k] = sum_e fc_w[e][k] * mu_w[n][e]  (f16, K padded with zeros)
//       bc[n]    = sum_e fc_b[e]   * mu_w[n][e] + mu_b[n]
//       block 0 zeroes counts[0..63] only (64 threads; e_sum, n_valid not zeroed)
// ---------------------------------------------------------------------------
__global__ __launch_bounds__(64)
void prep_kernel(const float* __restrict__ fc_w, const float* __restrict__ fc_b,
                 const float* __restrict__ mu_w, const float* __restrict__ mu_b,
                 _Float16* __restrict__ Wc, float* __restrict__ bc,
                 float* __restrict__ acc)
{
    __shared__ float sMw[EMB];
    const int n = blockIdx.x;      // output column 0..127
    const int t = threadIdx.x;     // 0..63

    for (int e = t; e < EMB; e += 64) sMw[e] = mu_w[n * EMB + e];
    __syncthreads();

    for (int k = t; k < KPAD; k += 64) {
        float s = 0.f;
        if (k < FEAT) {
            for (int e = 0; e < EMB; ++e)
                s = fmaf(fc_w[e * FEAT + k], sMw[e], s);
        }
        Wc[n * KPAD + k] = (_Float16)s;
    }
    if (t == 0) {
        float s = mu_b[n];
        for (int e = 0; e < EMB; ++e) s = fmaf(fc_b[e], sMw[e], s);
        bc[n] = s;
    }
    if (n == 0 && t < 66) acc[t] = 0.f;   // counts[0..63] only; t < 64, so acc[64..65] stay unzeroed
}

// ---------------------------------------------------------------------------
// fused: conv+relu+pool -> F(f16,LDS) -> WMMA GEMM mu = F @ Wc + bc
//        -> VQ argmin -> z store, loss/count atomics
// ---------------------------------------------------------------------------
__global__ __launch_bounds__(256)
void encoder_vq_kernel(const float* __restrict__ pr,
                       const int*   __restrict__ padmask,
                       const float* __restrict__ conv_w,
                       const float* __restrict__ conv_b,
                       const float* __restrict__ codebook,
                       const _Float16* __restrict__ Wc,
                       const float* __restrict__ bc,
                       float* __restrict__ acc,
                       float* __restrict__ zout)
{
    __shared__ _Float16 sF[ROWS * FSTRIDE];   // feature tile (A matrix), 41984 B
    __shared__ float    sS[ROWS * MUSTR];     // pr tile, then f32 mu, 33792 B
    __shared__ float    sBd[256];
    __shared__ int      sBi[256];
    __shared__ int      sIdx[ROWS];
    __shared__ int      sVal[ROWS];
    __shared__ float    sRed[256];

    const int tid  = threadIdx.x;
    const int row0 = blockIdx.x * ROWS;

    // warm codebook into caches (global_prefetch_b8)
    __builtin_prefetch(codebook + (size_t)(tid & (KCB - 1)) * ZD, 0, 3);

    // ---- stage pr tile into LDS ----
    for (int i = tid; i < ROWS * LC; i += 256) {
        int r = i >> 7, c = i & (LC - 1);
        sS[r * LC + c] = pr[(size_t)(row0 + r) * LC + c];
    }
    __syncthreads();

    // ---- conv(k=12) + relu + maxpool(4) -> f16 features ----
    for (int i = tid; i < ROWS * FEAT; i += 256) {
        int r  = i / FEAT, f = i - r * FEAT;
        int ch = f / 29,   p = f - ch * 29;
        float w[12];
        #pragma unroll
        for (int j = 0; j < 12; ++j) w[j] = conv_w[ch * 12 + j];
        const float bias = conv_b[ch];
        float m = 0.f;                           // relu floor (max(relu)=relu(max))
        const int t0 = p * 4;
        #pragma unroll
        for (int s = 0; s < 4; ++s) {
            float a = bias;
            #pragma unroll
            for (int j = 0; j < 12; ++j)
                a = fmaf(sS[r * LC + t0 + s + j], w[j], a);
            m = fmaxf(m, a);
        }
        sF[r * FSTRIDE + f] = (_Float16)m;
    }
    // zero K padding 290..327
    for (int i = tid; i < ROWS * (FSTRIDE - FEAT); i += 256) {
        int r = i / (FSTRIDE - FEAT);
        int k = FEAT + (i - r * (FSTRIDE - FEAT));
        sF[r * FSTRIDE + k] = (_Float16)0.f;
    }
    __syncthreads();

    // ---- WMMA GEMM: mu[64][128] = F[64][320] @ Wc[320][128] + bc ----
    const int lane = tid & 31;
    const int wv   = tid >> 5;        // 8 waves
    const int rt   = wv & 3;          // row tile 0..3
    const int g    = wv >> 2;         // col-tile group 0..1 (4 tiles each)
    const int ln16 = lane & 15;
    const int hi   = lane >> 4;       // lane half: selects K half, k = 16*hi + s (contiguous order)

    v8f accv[4] = {};
    const _Float16* aBase = &sF[(rt * 16 + ln16) * FSTRIDE];

    for (int kb = 0; kb < KPAD; kb += 32) {
        union { v16h v; v8h h[2]; } A;
        A.h[0] = *(const v8h*)(aBase + kb + hi * 16);       // K = kb+{0..7}/{16..23}
        A.h[1] = *(const v8h*)(aBase + kb + hi * 16 + 8);   // K = kb+{8..15}/{24..31}
        #pragma unroll
        for (int t = 0; t < 4; ++t) {
            const int col = (g * 4 + t) * 16 + ln16;
            const _Float16* bp = Wc + (size_t)col * KPAD + kb + hi * 16;
            union { v16h v; v8h h[2]; } B;
            B.h[0] = *(const v8h*)(bp);
            B.h[1] = *(const v8h*)(bp + 8);
            accv[t] = __builtin_amdgcn_wmma_f32_16x16x32_f16(
                false, A.v, false, B.v, (short)0, accv[t], false, false);
        }
    }
    // bias + spill f32 mu to LDS (C layout: VGPR j -> M = rt*16 + hi*8 + j, N = ln16)
    #pragma unroll
    for (int t = 0; t < 4; ++t) {
        const int col = (g * 4 + t) * 16 + ln16;
        const float b = bc[col];
        #pragma unroll
        for (int j = 0; j < 8; ++j) {
            const int mrow = rt * 16 + hi * 8 + j;
            sS[mrow * MUSTR + col] = accv[t][j] + b;
        }
    }
    __syncthreads();

    // ---- VQ: per-row argmin over 64 codes (f32 VALU for exactness) ----
    {
        const int vr = tid >> 2;          // row 0..63
        const int vq = tid & 3;           // 16-code quarter
        float best = 3.4e38f; int bidx = vq * 16;
        for (int c = vq * 16; c < vq * 16 + 16; ++c) {
            const float* cb = codebook + (size_t)c * ZD;
            float d = 0.f;
            for (int k = 0; k < ZD; ++k) {
                float df = sS[vr * MUSTR + k] - cb[k];
                d = fmaf(df, df, d);
            }
            if (d < best) { best = d; bidx = c; }   // first-min tie break (ascending c)
        }
        sBd[tid] = best; sBi[tid] = bidx;
    }
    __syncthreads();
    if (tid < ROWS) {
        float b0 = sBd[tid * 4]; int i0 = sBi[tid * 4];
        #pragma unroll
        for (int q = 1; q < 4; ++q) {
            float bq = sBd[tid * 4 + q];
            if (bq < b0) { b0 = bq; i0 = sBi[tid * 4 + q]; }
        }
        sIdx[tid] = i0;
        const int valid = (padmask[row0 + tid] == 0) ? 1 : 0;
        sVal[tid] = valid;
        if (valid) atomicAdd(&acc[i0], 1.0f);       // code usage counts
    }
    __syncthreads();

    // ---- z = codebook[idx] (STE forward value), commitment-loss partials ----
    float esum = 0.f;
    for (int i = tid; i < ROWS * ZD; i += 256) {
        int r = i >> 7, k = i & (ZD - 1);
        float qv = codebook[(size_t)sIdx[r] * ZD + k];
        zout[(size_t)(row0 + r) * ZD + k] = qv;
        if (sVal[r]) {
            float df = qv - sS[r * MUSTR + k];
            esum = fmaf(df, df, esum);
        }
    }
    sRed[tid] = esum;
    __syncthreads();
    for (int s = 128; s > 0; s >>= 1) {
        if (tid < s) sRed[tid] += sRed[tid + s];
        __syncthreads();
    }
    if (tid == 0) {
        atomicAdd(&acc[64], sRed[0]);               // sum |q-mu|^2 over valid rows
        float cnt = 0.f;
        for (int r = 0; r < ROWS; ++r) cnt += (float)sVal[r];
        atomicAdd(&acc[65], cnt);                   // n_valid
    }
}

// ---------------------------------------------------------------------------
// finalize: cmt_loss and perplexity scalars
// ---------------------------------------------------------------------------
__global__ __launch_bounds__(64)
void finalize_kernel(const float* __restrict__ acc, float* __restrict__ out)
{
    if (threadIdx.x == 0 && blockIdx.x == 0) {
        float nv = acc[65];
        if (nv < 1.f) nv = 1.f;
        float ent = 0.f;
        for (int k = 0; k < KCB; ++k) {
            float p = acc[k] / nv;
            ent += p * __logf(p + 1e-10f);
        }
        out[0] = 0.25f * acc[64] / (nv * (float)ZD);  // cmt_loss
        out[1] = __expf(-ent);                        // perplexity
    }
}

// ---------------------------------------------------------------------------
extern "C" void kernel_launch(void* const* d_in, const int* in_sizes, int n_in,
                              void* d_out, int out_size, void* d_ws, size_t ws_size,
                              hipStream_t stream)
{
    const float* pr     = (const float*)d_in[0];
    const int*   mask   = (const int*)  d_in[1];   // track_pad_mask (bool)
    const float* conv_w = (const float*)d_in[2];
    const float* conv_b = (const float*)d_in[3];
    const float* fc_w   = (const float*)d_in[4];
    const float* fc_b   = (const float*)d_in[5];
    const float* mu_w   = (const float*)d_in[6];
    const float* mu_b   = (const float*)d_in[7];
    const float* cb     = (const float*)d_in[8];

    char* ws = (char*)d_ws;
    _Float16* Wc  = (_Float16*)ws;                      // 128*320*2 = 81920 B
    float*    bc  = (float*)(ws + 81920);               // 128 f32
    float*    acc = (float*)(ws + 81920 + 512);         // counts[64], e_sum, n_valid

    const int bs = in_sizes[1];                         // BS = 131072
    float* zout  = (float*)d_out;                       // [bs, 128]
    float* stats = zout + (size_t)bs * ZD;              // cmt_loss, perplexity

    hipLaunchKernelGGL(prep_kernel, dim3(ZD), dim3(64), 0, stream,
                       fc_w, fc_b, mu_w, mu_b, Wc, bc, acc);
    hipLaunchKernelGGL(encoder_vq_kernel, dim3(bs / ROWS), dim3(256), 0, stream,
                       pr, mask, conv_w, conv_b, cb, Wc, bc, acc, zout);
    hipLaunchKernelGGL(finalize_kernel, dim3(1), dim3(64), 0, stream, acc, stats);
}
